// ArchNet_3539053052569
// MI455X (gfx1250) — compile-verified, matrix operand order checked by reading, not run on hardware
//
#include <hip/hip_runtime.h>
#include <hip/hip_bf16.h>

#define kN 100000   // nodes
#define kE 800000   // edges
#define kG 64       // graphs

typedef __attribute__((ext_vector_type(16))) __bf16 v16bf;
typedef __attribute__((ext_vector_type(8)))  __bf16 v8bf;
typedef __attribute__((ext_vector_type(8)))  float  v8f;

__device__ __forceinline__ void atomic_add_f32(float* p, float v) {
    __hip_atomic_fetch_add(p, v, __ATOMIC_RELAXED, __HIP_MEMORY_SCOPE_AGENT);
}

// ---------------- Layer 1 (d_in = 1): scalar scatter + elementwise ----------------
__global__ void k_scatter_scalar(const int* __restrict__ src, const int* __restrict__ dst,
                                 const float* __restrict__ x, float* __restrict__ agg) {
    unsigned e = blockIdx.x * blockDim.x + threadIdx.x;
    if (e >= kE) return;
    atomic_add_f32(&agg[dst[e]], x[src[e]]);
}

__global__ void k_layer1(const float* __restrict__ x, const float* __restrict__ agg,
                         const float* __restrict__ Wr, const float* __restrict__ br,
                         const float* __restrict__ Wo, float* __restrict__ h1) {
    unsigned t = blockIdx.x * blockDim.x + threadIdx.x;
    if (t >= kN * 128u) return;
    unsigned i = t >> 7, j = t & 127u;
    float v = fmaf(agg[i], Wr[j], fmaf(x[i], Wo[j], br[j]));
    h1[t] = fmaxf(v, 0.0f);
}

// ---------------- Edge scatter-add of K-wide features (4 floats / thread) ----------------
__global__ void k_scatter_vec(const int* __restrict__ src, const int* __restrict__ dst,
                              const float* __restrict__ h, float* __restrict__ agg, int K) {
    unsigned t = blockIdx.x * blockDim.x + threadIdx.x;
    unsigned chunks = (unsigned)K >> 2;
    unsigned e = t / chunks;
    if (e >= kE) return;
    unsigned c = (t - e * chunks) << 2;
    const float4 v = *(const float4*)(h + (size_t)src[e] * K + c);
    float* p = agg + (size_t)dst[e] * K + c;
    atomic_add_f32(p + 0, v.x);
    atomic_add_f32(p + 1, v.y);
    atomic_add_f32(p + 2, v.z);
    atomic_add_f32(p + 3, v.w);
}

// ---------------- Pack A = [agg | h] as bf16 row-major [N, 2K] ----------------
__global__ void k_packA(const float* __restrict__ agg, const float* __restrict__ h,
                        __bf16* __restrict__ Ain, int K) {
    size_t t = (size_t)blockIdx.x * blockDim.x + threadIdx.x;
    int Ktot = 2 * K;
    size_t total = (size_t)kN * Ktot;
    if (t >= total) return;
    size_t i = t / Ktot;
    int k = (int)(t - i * Ktot);
    float v = (k < K) ? agg[i * K + k] : h[i * K + (k - K)];
    Ain[t] = (__bf16)v;
}

// ---------------- Pack W^T = [W_rel; W_root] as bf16 [Nout, 2K] (K contiguous) ----------------
__global__ void k_packW(const float* __restrict__ Wrel, const float* __restrict__ Wroot,
                        __bf16* __restrict__ Wt, int K, int Nout) {
    size_t t = (size_t)blockIdx.x * blockDim.x + threadIdx.x;
    int Ktot = 2 * K;
    size_t total = (size_t)Nout * Ktot;
    if (t >= total) return;
    int n = (int)(t / Ktot);
    int k = (int)(t - (size_t)n * Ktot);
    float v = (k < K) ? Wrel[(size_t)k * Nout + n] : Wroot[(size_t)(k - K) * Nout + n];
    Wt[t] = (__bf16)v;
}

__global__ void k_counts(const int* __restrict__ batch, float* __restrict__ cnt) {
    unsigned i = blockIdx.x * blockDim.x + threadIdx.x;
    if (i >= kN) return;
    atomic_add_f32(&cnt[batch[i]], 1.0f);
}

// ---------------- WMMA GEMM: out[N,NOUT] = A[N,KTOT] @ Wt[NOUT,KTOT]^T + bias ----------------
// 8 waves/block, each wave owns one 16x16 tile; grid = (N/16, NOUT/128).
// POOL mode fuses segment-sum pooling (atomicAdd into pool[batch[row]][col]).
template <int KTOT, int NOUT, bool RELU, bool POOL>
__global__ __launch_bounds__(256) void k_gemm_wmma(
    const __bf16* __restrict__ A, const __bf16* __restrict__ Wt,
    const float* __restrict__ bias, float* __restrict__ out,
    const int* __restrict__ batch, float* __restrict__ pool) {
    const int wave = threadIdx.x >> 5;
    const int lane = threadIdx.x & 31;
    const int hi   = lane >> 4;   // lane group (0: lanes 0-15, 1: lanes 16-31)
    const int ln   = lane & 15;
    const int m0 = blockIdx.x * 16;
    const int n0 = (blockIdx.y * 8 + wave) * 16;

    // ISA 16-bit A layout: lane-lo holds K 0-7 & 16-23, lane-hi holds K 8-15 & 24-31,
    // each as two contiguous 16B runs -> two b128 loads per fragment.
    const __bf16* arow = A  + (size_t)(m0 + ln) * KTOT + hi * 8;
    const __bf16* brow = Wt + (size_t)(n0 + ln) * KTOT + hi * 8;

    v8f c = {};
#pragma unroll
    for (int k0 = 0; k0 < KTOT; k0 += 32) {
        v8bf a0 = *(const v8bf*)(arow + k0);
        v8bf a1 = *(const v8bf*)(arow + k0 + 16);
        v8bf b0 = *(const v8bf*)(brow + k0);
        v8bf b1 = *(const v8bf*)(brow + k0 + 16);
        v16bf a = __builtin_shufflevector(a0, a1, 0,1,2,3,4,5,6,7,8,9,10,11,12,13,14,15);
        v16bf b = __builtin_shufflevector(b0, b1, 0,1,2,3,4,5,6,7,8,9,10,11,12,13,14,15);
        c = __builtin_amdgcn_wmma_f32_16x16x32_bf16(false, a, false, b, (short)0, c, false, false);
    }

    const int col = n0 + ln;
    const float bv = bias[col];
    if constexpr (!POOL) {
#pragma unroll
        for (int g = 0; g < 8; ++g) {
            float v = c[g] + bv;
            if (RELU) v = fmaxf(v, 0.0f);
            out[(size_t)(m0 + g + hi * 8) * NOUT + col] = v;
        }
    } else {
#pragma unroll
        for (int g = 0; g < 8; ++g) {
            int r = m0 + g + hi * 8;
            atomic_add_f32(&pool[(size_t)batch[r] * NOUT + col], c[g] + bv);
        }
    }
}

// ---------------- Mean + L2 normalize per graph ----------------
__global__ void k_finalize(const float* __restrict__ pool, const float* __restrict__ cnt,
                           float* __restrict__ out) {
    __shared__ float red[256];
    int g = blockIdx.x;
    float c = fmaxf(cnt[g], 1.0f);
    float ss = 0.0f;
    for (int j = threadIdx.x; j < 512; j += 256) {
        float v = pool[g * 512 + j] / c;
        ss += v * v;
    }
    red[threadIdx.x] = ss;
    __syncthreads();
    for (int s = 128; s > 0; s >>= 1) {
        if (threadIdx.x < s) red[threadIdx.x] += red[threadIdx.x + s];
        __syncthreads();
    }
    float norm = fmaxf(sqrtf(red[0]), 1e-12f);
    for (int j = threadIdx.x; j < 512; j += 256) {
        out[g * 512 + j] = (pool[g * 512 + j] / c) / norm;
    }
}

extern "C" void kernel_launch(void* const* d_in, const int* in_sizes, int n_in,
                              void* d_out, int out_size, void* d_ws, size_t ws_size,
                              hipStream_t stream) {
    (void)in_sizes; (void)n_in; (void)out_size; (void)ws_size;

    const float* x     = (const float*)d_in[0];
    const int*   ei    = (const int*)d_in[1];
    const int*   src   = ei;
    const int*   dst   = ei + kE;
    const int*   batch = (const int*)d_in[2];
    const float* Wr1 = (const float*)d_in[3];
    const float* br1 = (const float*)d_in[4];
    const float* Wo1 = (const float*)d_in[5];
    const float* Wr2 = (const float*)d_in[6];
    const float* br2 = (const float*)d_in[7];
    const float* Wo2 = (const float*)d_in[8];
    const float* Wr3 = (const float*)d_in[9];
    const float* br3 = (const float*)d_in[10];
    const float* Wo3 = (const float*)d_in[11];

    char* ws = (char*)d_ws;
    size_t off = 0;
    auto alloc = [&](size_t bytes) -> char* {
        char* p = ws + off;
        off += (bytes + 255) & ~(size_t)255;
        return p;
    };
    float*  agg1 = (float*)alloc((size_t)kN * 4);
    float*  h1   = (float*)alloc((size_t)kN * 128 * 4);
    float*  agg2 = (float*)alloc((size_t)kN * 128 * 4);
    float*  h2   = (float*)alloc((size_t)kN * 256 * 4);
    float*  agg3 = (float*)alloc((size_t)kN * 256 * 4);
    __bf16* Ain2 = (__bf16*)alloc((size_t)kN * 256 * 2);
    __bf16* Ain3 = (__bf16*)alloc((size_t)kN * 512 * 2);
    __bf16* Wt2  = (__bf16*)alloc((size_t)256 * 256 * 2);
    __bf16* Wt3  = (__bf16*)alloc((size_t)512 * 512 * 2);
    float*  pool = (float*)alloc((size_t)kG * 512 * 4);
    float*  cnt  = (float*)alloc((size_t)kG * 4);

    // zero accumulation buffers (harness does not re-zero between replays)
    (void)hipMemsetAsync(agg1, 0, (size_t)kN * 4, stream);
    (void)hipMemsetAsync(agg2, 0, (size_t)kN * 128 * 4, stream);
    (void)hipMemsetAsync(agg3, 0, (size_t)kN * 256 * 4, stream);
    (void)hipMemsetAsync(pool, 0, (size_t)kG * 512 * 4, stream);
    (void)hipMemsetAsync(cnt,  0, (size_t)kG * 4, stream);

    // one-time-per-call weight repack (tiny vs the reuse in 6250 row tiles)
    {
        size_t t2 = (size_t)256 * 256;
        k_packW<<<(unsigned)((t2 + 255) / 256), 256, 0, stream>>>(Wr2, Wo2, Wt2, 128, 256);
        size_t t3 = (size_t)512 * 512;
        k_packW<<<(unsigned)((t3 + 255) / 256), 256, 0, stream>>>(Wr3, Wo3, Wt3, 256, 512);
    }
    k_counts<<<(kN + 255) / 256, 256, 0, stream>>>(batch, cnt);

    // Layer 1 (rank-1, elementwise)
    k_scatter_scalar<<<(kE + 255) / 256, 256, 0, stream>>>(src, dst, x, agg1);
    k_layer1<<<(kN * 128 + 255) / 256, 256, 0, stream>>>(x, agg1, Wr1, br1, Wo1, h1);

    // Layer 2: scatter -> pack -> WMMA GEMM (+bias+relu)
    {
        size_t t = (size_t)kE * 32;
        k_scatter_vec<<<(unsigned)((t + 255) / 256), 256, 0, stream>>>(src, dst, h1, agg2, 128);
    }
    {
        size_t t = (size_t)kN * 256;
        k_packA<<<(unsigned)((t + 255) / 256), 256, 0, stream>>>(agg2, h1, Ain2, 128);
    }
    k_gemm_wmma<256, 256, true, false><<<dim3(kN / 16, 2), 256, 0, stream>>>(
        Ain2, Wt2, br2, h2, nullptr, nullptr);

    // Layer 3: scatter -> pack -> WMMA GEMM with fused segment-sum pooling
    {
        size_t t = (size_t)kE * 64;
        k_scatter_vec<<<(unsigned)((t + 255) / 256), 256, 0, stream>>>(src, dst, h2, agg3, 256);
    }
    {
        size_t t = (size_t)kN * 512;
        k_packA<<<(unsigned)((t + 255) / 256), 256, 0, stream>>>(agg3, h2, Ain3, 256);
    }
    k_gemm_wmma<512, 512, false, true><<<dim3(kN / 16, 4), 256, 0, stream>>>(
        Ain3, Wt3, br3, nullptr, batch, pool);

    // mean pool + L2 normalize
    k_finalize<<<kG, 256, 0, stream>>>(pool, cnt, (float*)d_out);
}
